// GeneratedTreeClassifier_50062138802284
// MI455X (gfx1250) — matrix operand order repaired and checked by reading, not compiled, not run on hardware
//
#include <hip/hip_runtime.h>
#include <math.h>

#define INPUT_DIM   512
#define N_CLASSES   512
#define N_TREES     64
#define N_LEAVES    8
#define N_INTERNAL  7
#define PARAM_PER_TREE (N_INTERNAL * (INPUT_DIM + 1) + N_LEAVES * N_CLASSES) // 7687
#define BATCH       16384
#define DEC_COLS    256   // 64 trees * 4 used internal nodes
#define LEAF_COLS   512   // 64 trees * 8 leaves

typedef __attribute__((ext_vector_type(16))) __bf16        v16bf;
typedef __attribute__((ext_vector_type(8)))  float         v8f;
typedef __attribute__((ext_vector_type(4)))  unsigned int  u32x4;

union Frag16 { u32x4 q[2]; v16bf v; };

__device__ __forceinline__ unsigned short f2bf(float f) {
  unsigned int u = __float_as_uint(f);
  unsigned int r = (u + 0x7FFFu + ((u >> 16) & 1u)) >> 16; // RNE
  return (unsigned short)r;
}

// A fragment: 16x32 bf16, lane L (ln=L&15,hi=L>>4) holds row M=ln,
// elements 0..7 -> K = k0 + 8*hi + e ; elements 8..15 -> K = k0 + 16 + 8*hi + (e-8)
__device__ __forceinline__ v16bf load_a_frag(const unsigned short* __restrict__ A,
                                             int row, int k0, int hi) {
  Frag16 f;
  f.q[0] = *(const u32x4*)(A + (size_t)row * 512 + k0 + 8 * hi);
  f.q[1] = *(const u32x4*)(A + (size_t)row * 512 + k0 + 16 + 8 * hi);
  return f.v;
}

// B fragment: 32x16 bf16 from an N-major (transposed) matrix BT[n][k]:
// lane holds column N=ln, elements 0..7 -> K = k0 + 8*hi + e ; elements 8..15 -> K = k0 + 16 + 8*hi + (e-8)
__device__ __forceinline__ v16bf load_b_frag(const unsigned short* __restrict__ BT,
                                             int col, int k0, int hi) {
  Frag16 f;
  f.q[0] = *(const u32x4*)(BT + (size_t)col * 512 + k0 + 8 * hi);
  f.q[1] = *(const u32x4*)(BT + (size_t)col * 512 + k0 + 16 + 8 * hi);
  return f.v;
}

// ---------- prep kernels ----------

__global__ void k_cvt_x(const float* __restrict__ x, unsigned short* __restrict__ xb, int n) {
  int i = blockIdx.x * blockDim.x + threadIdx.x;
  int stride = gridDim.x * blockDim.x;
  for (; i < n; i += stride) xb[i] = f2bf(x[i]);
}

// WT[n][k] (n = tree*4 + node, only nodes 0..3 used), bias[n]
__global__ void k_prep_split(const float* __restrict__ tp,
                             unsigned short* __restrict__ wt,
                             float* __restrict__ bias) {
  int n = blockIdx.x;            // 0..255
  int t = n >> 2, i = n & 3;
  const float* base = tp + (size_t)t * PARAM_PER_TREE;
  const float* wrow = base + i * INPUT_DIM;
  for (int k = threadIdx.x; k < INPUT_DIM; k += blockDim.x)
    wt[(size_t)n * INPUT_DIM + k] = f2bf(wrow[k]);
  if (threadIdx.x == 0) bias[n] = base[N_INTERNAL * INPUT_DIM + i];
}

// MT[c][t*8+l] = tw[t] * softmax(leaf_logits[t,l])[c]   (N-major for B frags)
__global__ __launch_bounds__(256) void k_prep_leaf(const float* __restrict__ tp,
                                                   const float* __restrict__ tw,
                                                   unsigned short* __restrict__ mt) {
  __shared__ float red[256];
  const int row = blockIdx.x;    // t*8 + l, 0..511
  const int t = row >> 3, l = row & 7;
  const float* ll = tp + (size_t)t * PARAM_PER_TREE
                       + N_INTERNAL * INPUT_DIM + N_INTERNAL
                       + (size_t)l * N_CLASSES;
  const int tid = threadIdx.x;
  float m = -3.4e38f;
  for (int c = tid; c < N_CLASSES; c += 256) m = fmaxf(m, ll[c]);
  red[tid] = m; __syncthreads();
  for (int s = 128; s > 0; s >>= 1) {
    if (tid < s) red[tid] = fmaxf(red[tid], red[tid + s]);
    __syncthreads();
  }
  const float mx = red[0];
  __syncthreads();
  float sum = 0.f;
  for (int c = tid; c < N_CLASSES; c += 256) sum += __expf(ll[c] - mx);
  red[tid] = sum; __syncthreads();
  for (int s = 128; s > 0; s >>= 1) {
    if (tid < s) red[tid] += red[tid + s];
    __syncthreads();
  }
  const float inv = tw[t] / red[0];
  for (int c = tid; c < N_CLASSES; c += 256)
    mt[(size_t)c * LEAF_COLS + row] = f2bf(__expf(ll[c] - mx) * inv);
}

// ---------- GEMM 1: decisions + sigmoid + leaf probs (bf16 out) ----------
// block = 128 threads (4 waves); wave tile 16M x 64N; block tile 64M x 64N
__global__ __launch_bounds__(128) void k_dec_leaf(const unsigned short* __restrict__ Xb,
                                                  const unsigned short* __restrict__ WT,
                                                  const float* __restrict__ bias,
                                                  unsigned short* __restrict__ LP) {
  __shared__ float dtile[4][16][65];
  const int lane = threadIdx.x & 31;
  const int wave = threadIdx.x >> 5;
  const int hi = lane >> 4, ln = lane & 15;
  const int baseM = blockIdx.x * 64 + wave * 16;
  const int baseN = blockIdx.y * 64;            // decision columns
  v8f acc[4] = {};
  const int row = baseM + ln;
  for (int k0 = 0; k0 < INPUT_DIM; k0 += 32) {
    v16bf a = load_a_frag(Xb, row, k0, hi);
#pragma unroll
    for (int nt = 0; nt < 4; ++nt) {
      v16bf b = load_b_frag(WT, baseN + nt * 16 + ln, k0, hi);
      acc[nt] = __builtin_amdgcn_wmma_f32_16x16x32_bf16(
          false, a, false, b, (short)0, acc[nt], false, false);
    }
  }
  // C layout: VGPR j holds (M = j + 8*hi, N = ln); apply bias+sigmoid, stage to LDS
#pragma unroll
  for (int nt = 0; nt < 4; ++nt) {
    const float bv = bias[baseN + nt * 16 + ln];
#pragma unroll
    for (int j = 0; j < 8; ++j) {
      float z = acc[nt][j] + bv;
      float d = 1.0f / (1.0f + __expf(-z));
      dtile[wave][hi * 8 + j][nt * 16 + ln] = d;
    }
  }
  __syncthreads();
  // each lane: row r = ln, trees [hi*8, hi*8+8) of this wave's 16-tree strip
  const int r = ln;
  const int rowg = baseM + r;
#pragma unroll
  for (int tt = 0; tt < 8; ++tt) {
    int lt = hi * 8 + tt;                        // local tree 0..15
    float d0 = dtile[wave][r][lt * 4 + 0];
    float d1 = dtile[wave][r][lt * 4 + 1];
    float d2 = dtile[wave][r][lt * 4 + 2];
    float d3 = dtile[wave][r][lt * 4 + 3];
    float inv = 1.0f / (4.0f + d3 + 1e-8f);      // sum of leaf probs is exactly 4 + d3
    unsigned short o0 = f2bf(inv);               // leaf 0 == 1 * inv
    unsigned short o1 = f2bf(d0 * inv);
    unsigned short o2 = f2bf((1.0f - d0) * inv);
    unsigned short o3 = f2bf(d1 * inv);
    unsigned short o4 = f2bf((1.0f - d1) * inv);
    unsigned short o5 = f2bf(d2 * inv);
    unsigned short o6 = f2bf((1.0f - d2) * inv);
    unsigned short o7 = f2bf(d3 * inv);
    u32x4 q;
    q.x = (unsigned)o0 | ((unsigned)o1 << 16);
    q.y = (unsigned)o2 | ((unsigned)o3 << 16);
    q.z = (unsigned)o4 | ((unsigned)o5 << 16);
    q.w = (unsigned)o6 | ((unsigned)o7 << 16);
    *(u32x4*)(LP + (size_t)rowg * LEAF_COLS + (blockIdx.y * 16 + lt) * 8) = q;
  }
}

// ---------- GEMM 2: out = LP(16384x512) @ M(512x512), fp32 out ----------
__global__ __launch_bounds__(128) void k_out(const unsigned short* __restrict__ LP,
                                             const unsigned short* __restrict__ MT,
                                             float* __restrict__ out) {
  const int lane = threadIdx.x & 31;
  const int wave = threadIdx.x >> 5;
  const int hi = lane >> 4, ln = lane & 15;
  const int baseM = blockIdx.x * 64 + wave * 16;
  const int baseN = blockIdx.y * 64;
  v8f acc[4] = {};
  const int row = baseM + ln;
  for (int k0 = 0; k0 < LEAF_COLS; k0 += 32) {
    v16bf a = load_a_frag(LP, row, k0, hi);
#pragma unroll
    for (int nt = 0; nt < 4; ++nt) {
      v16bf b = load_b_frag(MT, baseN + nt * 16 + ln, k0, hi);
      acc[nt] = __builtin_amdgcn_wmma_f32_16x16x32_bf16(
          false, a, false, b, (short)0, acc[nt], false, false);
    }
  }
#pragma unroll
  for (int nt = 0; nt < 4; ++nt)
#pragma unroll
    for (int j = 0; j < 8; ++j)
      out[(size_t)(baseM + hi * 8 + j) * N_CLASSES + baseN + nt * 16 + ln] = acc[nt][j];
}

extern "C" void kernel_launch(void* const* d_in, const int* in_sizes, int n_in,
                              void* d_out, int out_size, void* d_ws, size_t ws_size,
                              hipStream_t stream) {
  const float* x  = (const float*)d_in[0];   // (16384, 512)
  const float* tp = (const float*)d_in[1];   // (1, 64*7687)
  const float* tw = (const float*)d_in[2];   // (1, 64)
  float* out = (float*)d_out;                // (16384, 512)

  char* ws = (char*)d_ws;
  size_t off = 0;
  auto alloc = [&](size_t bytes) -> void* {
    void* p = ws + off;
    off = (off + bytes + 255) & ~(size_t)255;
    return p;
  };
  unsigned short* Xb   = (unsigned short*)alloc((size_t)BATCH * INPUT_DIM * 2); // 16 MB
  unsigned short* LP   = (unsigned short*)alloc((size_t)BATCH * LEAF_COLS * 2); // 16 MB
  unsigned short* WT   = (unsigned short*)alloc((size_t)DEC_COLS * INPUT_DIM * 2);
  unsigned short* MT   = (unsigned short*)alloc((size_t)N_CLASSES * LEAF_COLS * 2);
  float*          bias = (float*)alloc(DEC_COLS * sizeof(float));

  k_cvt_x<<<2048, 256, 0, stream>>>(x, Xb, BATCH * INPUT_DIM);
  k_prep_split<<<DEC_COLS, 128, 0, stream>>>(tp, WT, bias);
  k_prep_leaf<<<LEAF_COLS, 256, 0, stream>>>(tp, tw, MT);

  dim3 g1(BATCH / 64, DEC_COLS / 64);   // (256, 4)
  k_dec_leaf<<<g1, 128, 0, stream>>>(Xb, WT, bias, LP);

  dim3 g2(BATCH / 64, N_CLASSES / 64);  // (256, 8)
  k_out<<<g2, 128, 0, stream>>>(LP, MT, out);
}
